// Attention_24300924960951
// MI455X (gfx1250) — matrix operand order repaired and checked by reading, not compiled, not run on hardware
//
#include <hip/hip_runtime.h>

typedef __attribute__((ext_vector_type(16))) _Float16 v16h;
typedef __attribute__((ext_vector_type(8)))  _Float16 v8h;
typedef __attribute__((ext_vector_type(4)))  _Float16 v4h;
typedef __attribute__((ext_vector_type(8)))  float    v8f;

namespace {
constexpr int kB   = 4;
constexpr int kS   = 4096;
constexpr int kD   = 64;
constexpr int kBQ  = 64;   // query rows per workgroup (4 waves x 16)
constexpr int kBK  = 32;   // keys per inner block
constexpr int kLDK = 72;   // halves: K tile row stride (rows 16B-aligned)
constexpr int kLDV = 36;   // halves: transposed-V row stride (conflict-free, 8B-aligned)
constexpr int kLDP = 40;   // halves: P scratch row stride (chunks 16B-aligned)
}

__device__ __forceinline__ float fast_exp2(float x) {
    return __builtin_amdgcn_exp2f(x);      // v_exp_f32 (native exp2)
}

// raw v_max_num: skip LLVM's fcanonicalize (our data has no sNaNs)
__device__ __forceinline__ float max_raw(float a, float b) {
    float d;
    asm("v_max_num_f32 %0, %1, %2" : "=v"(d) : "v"(a), "v"(b));
    return d;
}

// One butterfly round over all 8 rows, schedule pinned:
// 8x ds_swizzle -> one s_wait_dscnt -> 8x v_max_num. IMMS = xor<<10 | 0x1f.
#define SWZ_MAX_ROUND(v, IMMS)                                              \
    do {                                                                    \
        float t0, t1, t2, t3, t4, t5, t6, t7;                               \
        asm volatile(                                                       \
            "ds_swizzle_b32 %8,  %0 offset:" IMMS "\n\t"                    \
            "ds_swizzle_b32 %9,  %1 offset:" IMMS "\n\t"                    \
            "ds_swizzle_b32 %10, %2 offset:" IMMS "\n\t"                    \
            "ds_swizzle_b32 %11, %3 offset:" IMMS "\n\t"                    \
            "ds_swizzle_b32 %12, %4 offset:" IMMS "\n\t"                    \
            "ds_swizzle_b32 %13, %5 offset:" IMMS "\n\t"                    \
            "ds_swizzle_b32 %14, %6 offset:" IMMS "\n\t"                    \
            "ds_swizzle_b32 %15, %7 offset:" IMMS "\n\t"                    \
            "s_wait_dscnt 0x0\n\t"                                          \
            "v_max_num_f32 %0, %0, %8\n\t"                                  \
            "v_max_num_f32 %1, %1, %9\n\t"                                  \
            "v_max_num_f32 %2, %2, %10\n\t"                                 \
            "v_max_num_f32 %3, %3, %11\n\t"                                 \
            "v_max_num_f32 %4, %4, %12\n\t"                                 \
            "v_max_num_f32 %5, %5, %13\n\t"                                 \
            "v_max_num_f32 %6, %6, %14\n\t"                                 \
            "v_max_num_f32 %7, %7, %15"                                     \
            : "+v"(v[0]), "+v"(v[1]), "+v"(v[2]), "+v"(v[3]),               \
              "+v"(v[4]), "+v"(v[5]), "+v"(v[6]), "+v"(v[7]),               \
              "=&v"(t0), "=&v"(t1), "=&v"(t2), "=&v"(t3),                   \
              "=&v"(t4), "=&v"(t5), "=&v"(t6), "=&v"(t7));                  \
    } while (0)

template <int IMM>
__device__ __forceinline__ void swz_add_round(float (&v)[8]) {
    int t[8];
    #pragma unroll
    for (int r = 0; r < 8; ++r)
        t[r] = __builtin_amdgcn_ds_swizzle(__float_as_int(v[r]), IMM);
    #pragma unroll
    for (int r = 0; r < 8; ++r)
        v[r] = v[r] + __int_as_float(t[r]);
}

__launch_bounds__(128, 1)
__global__ void fa_wmma_kernel(const float* __restrict__ Q,
                               const float* __restrict__ K,
                               const float* __restrict__ V,
                               const int*   __restrict__ AM,
                               float*       __restrict__ O)
{
    __shared__ _Float16 sK [kBK * kLDK];     // K block: [key][feature]      (f16)
    __shared__ _Float16 sVt[kD  * kLDV];     // V block transposed: [feat][key]
    __shared__ _Float16 sP [4 * 16 * kLDP];  // per-wave P relayout scratch

    const int tid  = threadIdx.x;
    const int wave = tid >> 5;
    const int lane = tid & 31;
    const int m    = lane & 15;   // A-frag row / B,C,D-frag column
    const int hi   = lane >> 4;   // half-wave group

    const int b     = blockIdx.y;
    const int q0    = blockIdx.x * kBQ;
    const int qrow0 = q0 + wave * 16;
    const int qmaxw = qrow0 + 15;            // last query row of this wave

    // fold 1/sqrt(64) and log2(e) into Q so softmax uses native exp2
    const float qscale = 0.125f * 1.44269504088896340736f;

    // ---- Q tile -> two K=32 A fragments (f16), resident for whole loop
    v16h qa[2];
    {
        const float* qbase = Q + ((size_t)b * kS + (qrow0 + m)) * kD;
        #pragma unroll
        for (int c = 0; c < 2; ++c) {
            #pragma unroll
            for (int e = 0; e < 16; ++e) {
                const int k = 32 * c + 16 * hi + e;
                qa[c][e] = (_Float16)(qbase[k] * qscale);
            }
        }
    }

    v8f accO[4];
    float mrow[8], lrow[8];
    #pragma unroll
    for (int nt = 0; nt < 4; ++nt)
        #pragma unroll
        for (int r = 0; r < 8; ++r) accO[nt][r] = 0.0f;
    #pragma unroll
    for (int r = 0; r < 8; ++r) { mrow[r] = -INFINITY; lrow[r] = 0.0f; }

    _Float16* pb = &sP[wave * 16 * kLDP];
    const int vcol = tid & 63;               // feature column this thread stages
    const int vrg  = tid >> 6;               // key half (rows 0-15 / 16-31)

    const int nblk = (q0 + kBQ) / kBK;
    for (int blk = 0; blk < nblk; ++blk) {
        const int k0 = blk * kBK;

        // ---- issue ALL staging global loads first (overlap the barrier)
        const float4* Kg = (const float4*)(K + ((size_t)b * kS + k0) * kD);
        float4 kreg[4];
        #pragma unroll
        for (int i = 0; i < 4; ++i) kreg[i] = Kg[tid + 128 * i];

        const float* Vg = V + ((size_t)b * kS + k0 + vrg * 16) * kD + vcol;
        float vreg[16];
        #pragma unroll
        for (int e = 0; e < 16; ++e) vreg[e] = Vg[e * kD];  // lanes sweep cols: coalesced

        const int am0 = AM[(size_t)b * kS + k0 + m];
        const int am1 = AM[(size_t)b * kS + k0 + 16 + m];

        __syncthreads();                     // previous block fully consumed

        // ---- convert + store to LDS (K row-major, V transposed)
        #pragma unroll
        for (int i = 0; i < 4; ++i) {
            const int j = tid + 128 * i;     // float4 chunk id
            const int row = j >> 4;
            const int c4  = j & 15;
            v4h h;
            h[0] = (_Float16)kreg[i].x; h[1] = (_Float16)kreg[i].y;
            h[2] = (_Float16)kreg[i].z; h[3] = (_Float16)kreg[i].w;
            *(v4h*)&sK[row * kLDK + c4 * 4] = h;
        }
        #pragma unroll
        for (int cch = 0; cch < 4; ++cch) {
            v4h h;
            #pragma unroll
            for (int u = 0; u < 4; ++u) h[u] = (_Float16)vreg[cch * 4 + u];
            *(v4h*)&sVt[vcol * kLDV + vrg * 16 + cch * 4] = h;
        }
        __syncthreads();

        // ---- skip blocks entirely above this wave's causal frontier
        if (k0 <= qmaxw) {
            // load ALL four K B-fragments first, then issue the 4 score WMMAs
            v16h bf[2][2];
            #pragma unroll
            for (int t = 0; t < 2; ++t)
                #pragma unroll
                for (int c = 0; c < 2; ++c) {
                    const _Float16* kb = &sK[(16 * t + m) * kLDK + 32 * c + 16 * hi];
                    #pragma unroll
                    for (int e = 0; e < 16; ++e) bf[t][c][e] = kb[e];
                }

            v8f sc[2];
            #pragma unroll
            for (int t = 0; t < 2; ++t) {
                v8f acc = {};
                acc = __builtin_amdgcn_wmma_f32_16x16x32_f16(
                    false, qa[0], false, bf[t][0], (short)0, acc, false, false);
                acc = __builtin_amdgcn_wmma_f32_16x16x32_f16(
                    false, qa[1], false, bf[t][1], (short)0, acc, false, false);
                sc[t] = acc;
            }

            // issue V B-fragment loads now; latency hides under softmax VALU
            v16h bv[4];
            #pragma unroll
            for (int nt = 0; nt < 4; ++nt) {
                const _Float16* vb = &sVt[(16 * nt + m) * kLDV + 16 * hi];
                #pragma unroll
                for (int e = 0; e < 16; ++e) bv[nt][e] = vb[e];
            }

            // row max: 4 pinned butterfly rounds (8 swizzles, 1 wait, 8 maxes)
            float vmax[8];
            #pragma unroll
            for (int r = 0; r < 8; ++r) vmax[r] = max_raw(sc[0][r], sc[1][r]);
            SWZ_MAX_ROUND(vmax, "1055");   // xor 1  (0x041f)
            SWZ_MAX_ROUND(vmax, "2079");   // xor 2  (0x081f)
            SWZ_MAX_ROUND(vmax, "4127");   // xor 4  (0x101f)
            SWZ_MAX_ROUND(vmax, "8223");   // xor 8  (0x201f)

            // online softmax; padding/causal mask applied to P, not S
            float p0[8], p1[8], alpha[8];
            if (k0 + kBK - 1 <= qrow0) {
                // interior block: no causal masking for any row of this wave
                #pragma unroll
                for (int r = 0; r < 8; ++r) {
                    const float mnew = max_raw(mrow[r], vmax[r]);
                    const float a = fast_exp2(mrow[r] - mnew);
                    const float e0 = (am0 != 0) ? fast_exp2(sc[0][r] - mnew) : 0.0f;
                    const float e1 = (am1 != 0) ? fast_exp2(sc[1][r] - mnew) : 0.0f;
                    lrow[r] = lrow[r] * a + (e0 + e1);
                    mrow[r] = mnew;
                    alpha[r] = a;
                    p0[r] = e0;
                    p1[r] = e1;
                }
            } else {
                // diagonal block: full causal + padding mask
                #pragma unroll
                for (int r = 0; r < 8; ++r) {
                    const float mnew = max_raw(mrow[r], vmax[r]);
                    const float a = fast_exp2(mrow[r] - mnew);
                    const int   q = qrow0 + r + 8 * hi;
                    const bool ok0 = (am0 != 0) && (k0 + m      <= q);
                    const bool ok1 = (am1 != 0) && (k0 + 16 + m <= q);
                    const float e0 = ok0 ? fast_exp2(sc[0][r] - mnew) : 0.0f;
                    const float e1 = ok1 ? fast_exp2(sc[1][r] - mnew) : 0.0f;
                    lrow[r] = lrow[r] * a + (e0 + e1);
                    mrow[r] = mnew;
                    alpha[r] = a;
                    p0[r] = e0;
                    p1[r] = e1;
                }
            }

            #pragma unroll
            for (int nt = 0; nt < 4; ++nt)
                #pragma unroll
                for (int r = 0; r < 8; ++r) accO[nt][r] *= alpha[r];

            // P: C/D layout -> A-fragment layout via per-wave LDS scratch
            #pragma unroll
            for (int r = 0; r < 8; ++r) {
                const int mr = r + 8 * hi;
                pb[mr * kLDP + m]      = (_Float16)p0[r];
                pb[mr * kLDP + 16 + m] = (_Float16)p1[r];
            }
            asm volatile("s_wait_dscnt 0x0" ::: "memory");  // in-wave store->load order
            v16h pa;                                        // A frag: M=m, K=32 keys
            {
                const v8h lo = *(const v8h*)&pb[m * kLDP + 16 * hi];      // k=16hi+e
                const v8h hh = *(const v8h*)&pb[m * kLDP + 16 * hi + 8];  // k=16hi+8+e
                #pragma unroll
                for (int u = 0; u < 8; ++u) { pa[u] = lo[u]; pa[8 + u] = hh[u]; }
            }

            // O += P(16x32) x V(32x64): four N=16 tiles, operands already resident
            #pragma unroll
            for (int nt = 0; nt < 4; ++nt) {
                accO[nt] = __builtin_amdgcn_wmma_f32_16x16x32_f16(
                    false, pa, false, bv[nt], (short)0, accO[nt], false, false);
            }
        }
    }

    // ---- final row-sum reduction (deferred) + normalize + store
    {
        float lsum[8];
        #pragma unroll
        for (int r = 0; r < 8; ++r) lsum[r] = lrow[r];
        swz_add_round<0x041f>(lsum);
        swz_add_round<0x081f>(lsum);
        swz_add_round<0x101f>(lsum);
        swz_add_round<0x201f>(lsum);
        #pragma unroll
        for (int r = 0; r < 8; ++r) {
            const float inv = (lsum[r] > 0.0f) ? (1.0f / lsum[r]) : 0.0f;
            const int qrow = qrow0 + r + 8 * hi;
            float* ob = O + ((size_t)b * kS + qrow) * kD;
            #pragma unroll
            for (int nt = 0; nt < 4; ++nt)
                ob[16 * nt + m] = accO[nt][r] * inv;
        }
    }
}

extern "C" void kernel_launch(void* const* d_in, const int* in_sizes, int n_in,
                              void* d_out, int out_size, void* d_ws, size_t ws_size,
                              hipStream_t stream) {
    (void)in_sizes; (void)n_in; (void)out_size; (void)d_ws; (void)ws_size;
    const float* Q  = (const float*)d_in[0];
    const float* K  = (const float*)d_in[1];
    const float* V  = (const float*)d_in[2];
    const int*   AM = (const int*)d_in[3];
    float*       O  = (float*)d_out;

    dim3 grid(kS / kBQ, kB);   // (64, 4)
    dim3 block(128);
    fa_wmma_kernel<<<grid, block, 0, stream>>>(Q, K, V, AM, O);
}
